// DotProductAttention_14611478741280
// MI455X (gfx1250) — matrix operand order repaired and checked by reading, not compiled, not run on hardware
//
#include <hip/hip_runtime.h>
#include <hip/hip_bf16.h>

// ---------------------------------------------------------------------------
// Flash-attention for B=64, Lq=Lk=1024, d=64 (fp32 in/out), per-batch key mask.
// bf16 WMMA (v_wmma_f32_16x16x32_bf16) on gfx1250, wave32, 4 waves / block.
// Each wave owns 32 query rows (2 M-tiles) so every LDS B-fragment feeds
// two WMMAs (2x arithmetic intensity per ds byte vs 1 M-tile).
// ---------------------------------------------------------------------------

typedef __bf16 bf16_t;
typedef bf16_t v16bf __attribute__((ext_vector_type(16)));
typedef float  v8f   __attribute__((ext_vector_type(8)));

#define BATCH 64
#define SEQ   1024
#define DIM   64
#define KTILE 64          // keys per LDS tile
#define LDK   72          // padded LDS row stride (elements) for K/V tiles
#define QROWS_WAVE 32     // query rows per wave (2 WMMA M-tiles)
#define QROWS_BLK  128    // 4 waves
#define NEG_BIG (-1000000.0f)

__device__ __forceinline__ unsigned short f2bf(float f) {
    unsigned int u = __float_as_uint(f);
    unsigned int r = u + 0x7FFFu + ((u >> 16) & 1u);   // round-to-nearest-even
    return (unsigned short)(r >> 16);
}
__device__ __forceinline__ unsigned int pack2(float lo, float hi) {
    return (unsigned int)f2bf(lo) | ((unsigned int)f2bf(hi) << 16);
}

__global__ __launch_bounds__(128)
void flash_attn_f32_kernel(const float* __restrict__ Q,
                           const float* __restrict__ K,
                           const float* __restrict__ V,
                           const long long* __restrict__ vlen,
                           float* __restrict__ Out) {
    // LDS: K row-major [key][dd], V transposed [dd][key], P per-wave scratch.
    __shared__ __align__(16) unsigned short Ksh[KTILE * LDK];
    __shared__ __align__(16) unsigned short Vsh[DIM * LDK];
    __shared__ __align__(16) unsigned short Psh[4 * QROWS_WAVE * KTILE];

    const int b    = blockIdx.y;
    const int q0   = blockIdx.x * QROWS_BLK;
    const int tid  = threadIdx.x;
    const int wave = tid >> 5;
    const int lane = tid & 31;
    const int lm   = lane & 15;      // N / M index within fragment
    const int half = lane >> 4;      // lane half-group

    const int vl = (int)vlen[b];
    const float scale = 0.125f;      // 1/sqrt(64)

    // ---- Load Q A-fragments: 2 M-tiles x 2 K-halves ------------------------
    // 16-bit A layout: lane L (M = L%16), elems 0..15 -> K = 16*half + e
    // (contiguous, matching B; then +32 for second K-half).
    v16bf qa[2][2];
    #pragma unroll
    for (int mt = 0; mt < 2; ++mt) {
        const float* qrow =
            Q + ((size_t)b * SEQ + q0 + wave * QROWS_WAVE + 16 * mt + lm) * DIM;
        #pragma unroll
        for (int kf = 0; kf < 2; ++kf) {
            union { v16bf v; unsigned short u[16]; } fa;
            const int k1 = 32 * kf + 16 * half;
            const int k2 = k1 + 8;
            #pragma unroll
            for (int e = 0; e < 8; ++e) {
                fa.u[e]     = f2bf(qrow[k1 + e]);
                fa.u[8 + e] = f2bf(qrow[k2 + e]);
            }
            qa[mt][kf] = fa.v;
        }
    }

    // ---- Accumulators & online-softmax state ------------------------------
    v8f o[2][4];
    float mrow[2][8], lrow[2][8];
    #pragma unroll
    for (int mt = 0; mt < 2; ++mt) {
        #pragma unroll
        for (int c = 0; c < 4; ++c)
            o[mt][c] = (v8f){0.f, 0.f, 0.f, 0.f, 0.f, 0.f, 0.f, 0.f};
        #pragma unroll
        for (int r = 0; r < 8; ++r) { mrow[mt][r] = -3.0e38f; lrow[mt][r] = 0.f; }
    }

    unsigned short* Pw = &Psh[wave * QROWS_WAVE * KTILE];

    // ---- Key-tile loop ----------------------------------------------------
    for (int kt = 0; kt < SEQ / KTILE; ++kt) {
        const int k0 = kt * KTILE;

        __syncthreads();   // previous tile fully consumed before overwrite

        // Cooperative stage: 128 threads, each owns half a row (32 floats).
        {
            const int row = tid >> 1;
            const int cb  = (tid & 1) * 32;
            const float4* k4 = (const float4*)(K + ((size_t)b * SEQ + k0 + row) * DIM + cb);
            unsigned int* kdst = (unsigned int*)&Ksh[row * LDK + cb];
            #pragma unroll
            for (int e = 0; e < 8; ++e) {
                float4 x = k4[e];
                kdst[2 * e]     = pack2(x.x, x.y);
                kdst[2 * e + 1] = pack2(x.z, x.w);
            }
            const float4* v4 = (const float4*)(V + ((size_t)b * SEQ + k0 + row) * DIM + cb);
            #pragma unroll
            for (int e = 0; e < 8; ++e) {
                float4 x = v4[e];
                const int c = cb + 4 * e;
                Vsh[(c + 0) * LDK + row] = f2bf(x.x);
                Vsh[(c + 1) * LDK + row] = f2bf(x.y);
                Vsh[(c + 2) * LDK + row] = f2bf(x.z);
                Vsh[(c + 3) * LDK + row] = f2bf(x.w);
            }
            if (kt + 1 < SEQ / KTILE) {   // warm next tile toward L2
                __builtin_prefetch(K + ((size_t)b * SEQ + k0 + KTILE + row) * DIM + cb, 0, 1);
                __builtin_prefetch(V + ((size_t)b * SEQ + k0 + KTILE + row) * DIM + cb, 0, 1);
            }
        }
        __syncthreads();

        // ---- S = Q * K^T : each K B-fragment pair feeds BOTH M-tiles ------
        v8f s[2][4];
        #pragma unroll
        for (int n = 0; n < 4; ++n) {
            // B layout (32x16 bf16): lane L -> N = L%16 (key), elems e ->
            // K(feature) = 32*h + 16*half + e  (16 contiguous in LDS row)
            union { v16bf v; unsigned short u[16]; } fb0, fb1;
            const unsigned short* src0 = &Ksh[(16 * n + lm) * LDK + 16 * half];
            const unsigned short* src1 = src0 + 32;
            #pragma unroll
            for (int e = 0; e < 16; ++e) { fb0.u[e] = src0[e]; fb1.u[e] = src1[e]; }
            #pragma unroll
            for (int mt = 0; mt < 2; ++mt) {
                v8f c = (v8f){0.f, 0.f, 0.f, 0.f, 0.f, 0.f, 0.f, 0.f};
                c = __builtin_amdgcn_wmma_f32_16x16x32_bf16(
                        false, qa[mt][0], false, fb0.v, (short)0, c, false, false);
                c = __builtin_amdgcn_wmma_f32_16x16x32_bf16(
                        false, qa[mt][1], false, fb1.v, (short)0, c, false, false);
                s[mt][n] = c;
            }
        }

        // ---- Scale, mask, online softmax (per M-tile) ---------------------
        #pragma unroll
        for (int mt = 0; mt < 2; ++mt) {
            #pragma unroll
            for (int n = 0; n < 4; ++n) {
                const int col = k0 + 16 * n + lm;
                const bool keep = (col < vl);
                #pragma unroll
                for (int r = 0; r < 8; ++r) {
                    float vsc = s[mt][n][r] * scale;
                    s[mt][n][r] = keep ? vsc : NEG_BIG;
                }
            }
            #pragma unroll
            for (int r = 0; r < 8; ++r) {
                float t = fmaxf(fmaxf(s[mt][0][r], s[mt][1][r]),
                                fmaxf(s[mt][2][r], s[mt][3][r]));
                t = fmaxf(t, __shfl_xor(t, 1, 32));
                t = fmaxf(t, __shfl_xor(t, 2, 32));
                t = fmaxf(t, __shfl_xor(t, 4, 32));
                t = fmaxf(t, __shfl_xor(t, 8, 32));
                const float mnew = fmaxf(mrow[mt][r], t);
                const float corr = __expf(mrow[mt][r] - mnew);
                mrow[mt][r] = mnew;
                lrow[mt][r] *= corr;
                #pragma unroll
                for (int c = 0; c < 4; ++c) o[mt][c][r] *= corr;
                float rs = 0.f;
                #pragma unroll
                for (int n = 0; n < 4; ++n) {
                    float p = __expf(s[mt][n][r] - mnew);
                    s[mt][n][r] = p;
                    rs += p;
                }
                rs += __shfl_xor(rs, 1, 32);
                rs += __shfl_xor(rs, 2, 32);
                rs += __shfl_xor(rs, 4, 32);
                rs += __shfl_xor(rs, 8, 32);
                lrow[mt][r] += rs;
            }

            // Stage P (C-layout -> per-wave LDS, bf16)
            #pragma unroll
            for (int n = 0; n < 4; ++n)
                #pragma unroll
                for (int r = 0; r < 8; ++r)
                    Pw[(16 * mt + r + 8 * half) * KTILE + 16 * n + lm] =
                        f2bf(s[mt][n][r]);
        }

        // ---- Reload P as A-fragments --------------------------------------
        v16bf pa[2][2];
        #pragma unroll
        for (int mt = 0; mt < 2; ++mt) {
            const unsigned short* base = &Pw[(16 * mt + lm) * KTILE];
            #pragma unroll
            for (int kf = 0; kf < 2; ++kf) {
                union { v16bf v; unsigned short u[16]; } fa;
                const int k1 = 32 * kf + 16 * half;
                const int k2 = k1 + 8;
                #pragma unroll
                for (int e = 0; e < 8; ++e) {
                    fa.u[e]     = base[k1 + e];
                    fa.u[8 + e] = base[k2 + e];
                }
                pa[mt][kf] = fa.v;
            }
        }

        // ---- O += P * V : each V B-fragment pair feeds BOTH M-tiles -------
        #pragma unroll
        for (int c = 0; c < 4; ++c) {
            // B = V tile: lane L -> N = d-col 16*c + L%16,
            // elems e -> key = 32*h + 16*half + e (contiguous in Vsh row)
            union { v16bf v; unsigned short u[16]; } fb0, fb1;
            const unsigned short* src0 = &Vsh[(16 * c + lm) * LDK + 16 * half];
            const unsigned short* src1 = src0 + 32;
            #pragma unroll
            for (int e = 0; e < 16; ++e) { fb0.u[e] = src0[e]; fb1.u[e] = src1[e]; }
            #pragma unroll
            for (int mt = 0; mt < 2; ++mt) {
                o[mt][c] = __builtin_amdgcn_wmma_f32_16x16x32_bf16(
                               false, pa[mt][0], false, fb0.v, (short)0,
                               o[mt][c], false, false);
                o[mt][c] = __builtin_amdgcn_wmma_f32_16x16x32_bf16(
                               false, pa[mt][1], false, fb1.v, (short)0,
                               o[mt][c], false, false);
            }
        }
    }

    // ---- Epilogue: normalize & store fp32 ---------------------------------
    #pragma unroll
    for (int mt = 0; mt < 2; ++mt) {
        float inv[8];
        #pragma unroll
        for (int r = 0; r < 8; ++r) inv[r] = 1.0f / lrow[mt][r];
        #pragma unroll
        for (int c = 0; c < 4; ++c) {
            #pragma unroll
            for (int r = 0; r < 8; ++r) {
                const int row = q0 + wave * QROWS_WAVE + 16 * mt + r + 8 * half;
                Out[((size_t)b * SEQ + row) * DIM + 16 * c + lm] =
                    o[mt][c][r] * inv[r];
            }
        }
    }
}

extern "C" void kernel_launch(void* const* d_in, const int* in_sizes, int n_in,
                              void* d_out, int out_size, void* d_ws, size_t ws_size,
                              hipStream_t stream) {
    (void)in_sizes; (void)n_in; (void)out_size; (void)d_ws; (void)ws_size;
    const float*     Q  = (const float*)d_in[0];
    const float*     K  = (const float*)d_in[1];
    const float*     V  = (const float*)d_in[2];
    const long long* vl = (const long long*)d_in[3];
    float* Out = (float*)d_out;

    dim3 grid(SEQ / QROWS_BLK, BATCH);   // (8, 64): 128 query rows per block
    dim3 block(128);                     // 4 waves (wave32)
    flash_attn_f32_kernel<<<grid, block, 0, stream>>>(Q, K, V, vl, Out);
}
